// MolEmbedder_55490977464760
// MI455X (gfx1250) — matrix operand order repaired and checked by reading, not compiled, not run on hardware
//
#include <hip/hip_runtime.h>
#include <stdint.h>

#define NB 8
#define NN 256
#define ND 64
#define NE 64
#define NR 32

typedef __bf16 bf16_t;
typedef __attribute__((ext_vector_type(16))) __bf16 v16bf;
typedef __attribute__((ext_vector_type(8)))  float   v8f;

union FragU { v16bf v; uint4 q[2]; };

// Load a 16-element bf16 WMMA fragment as two contiguous 16B chunks from LDS.
static __device__ __forceinline__ v16bf lds_frag(const bf16_t* p0, const bf16_t* p1) {
  FragU f;
  f.q[0] = *(const uint4*)p0;
  f.q[1] = *(const uint4*)p1;
  return f.v;
}

// ---------------------------------------------------------------------------
// Kernel 1: node = LN(relu(emb[atom]) @ ne_w + ne_b); s = node@Ws; t = node@Wt
// ---------------------------------------------------------------------------
__global__ __launch_bounds__(64) void k_node(
    const int* __restrict__ atom, const float* __restrict__ emb,
    const float* __restrict__ ne_w, const float* __restrict__ ne_b,
    const float* __restrict__ ne_g, const float* __restrict__ ne_be,
    const float* __restrict__ ee_w1,
    float* __restrict__ node, float* __restrict__ s_out, float* __restrict__ t_out)
{
  __shared__ float er[64], nr[64], rb[64];
  const int row = blockIdx.x;
  const int d = threadIdx.x;
  const int a = atom[row];
  er[d] = fmaxf(emb[a*64 + d], 0.f);
  __syncthreads();
  float acc = ne_b[d];
  for (int k = 0; k < 64; ++k) acc += er[k] * ne_w[k*64 + d];
  rb[d] = acc; __syncthreads();
  for (int o = 32; o > 0; o >>= 1) { if (d < o) rb[d] += rb[d+o]; __syncthreads(); }
  const float mean = rb[0] * (1.0f/64.0f);
  __syncthreads();
  const float dv = acc - mean;
  rb[d] = dv*dv; __syncthreads();
  for (int o = 32; o > 0; o >>= 1) { if (d < o) rb[d] += rb[d+o]; __syncthreads(); }
  const float inv = rsqrtf(rb[0]*(1.0f/64.0f) + 1e-5f);
  const float nv = dv*inv*ne_g[d] + ne_be[d];
  nr[d] = nv;
  node[row*64 + d] = nv;
  __syncthreads();
  float as = 0.f, at = 0.f;
  for (int k = 0; k < 64; ++k) {
    const float nk = nr[k];
    as += nk * ee_w1[k*64 + d];          // Ws rows 0..63
    at += nk * ee_w1[(64 + k)*64 + d];   // Wt rows 64..127
  }
  s_out[row*64 + d] = as;
  t_out[row*64 + d] = at;
}

// ---------------------------------------------------------------------------
// Kernel 2: edge MLP via WMMA. One block per (b,i); waves own 16-j tiles.
//   h1 = relu(rbf@Wr + s_i + t_j + b1)   [WMMA 16x16x32 bf16, K=32]
//   h2 = relu(h1@W2 + b2)                [2 K-steps]
//   edge = LN(h2@W3 + b3)                [2 K-steps, LN via shfl_xor]
// ---------------------------------------------------------------------------
__global__ __launch_bounds__(256) void k_edge(
    const float* __restrict__ pos, const float* __restrict__ mask,
    const float* __restrict__ ee_w1, const float* __restrict__ b1,
    const float* __restrict__ w2, const float* __restrict__ b2,
    const float* __restrict__ w3, const float* __restrict__ b3,
    const float* __restrict__ g, const float* __restrict__ be,
    const float* __restrict__ s_in, const float* __restrict__ t_in,
    float* __restrict__ edge_out)
{
  __shared__ __align__(16) bf16_t wrT[64*32];        // Wr transposed: [n][k], k<32
  __shared__ __align__(16) bf16_t w2T[64*64];        // W2 transposed: [n][k]
  __shared__ __align__(16) bf16_t w3T[64*64];        // W3 transposed: [n][k]
  __shared__ __align__(16) bf16_t htile[8][16*64];   // per-wave activation tile
  __shared__ float posS[NN*3];
  __shared__ float maskS[NN];
  __shared__ float sS[64], b1S[64], b2S[64], b3S[64], gS[64], beS[64];

  const int bi = blockIdx.x;
  const int b  = bi >> 8;
  const int i  = bi & 255;
  const int tid = threadIdx.x;

  // Stage weights (bf16, pre-transposed so B-fragments are contiguous in K).
  for (int idx = tid; idx < 64*32; idx += 256) {
    const int n = idx & 63, k = idx >> 6;
    wrT[n*32 + k] = (bf16_t)ee_w1[(128 + k)*64 + n];   // Wr rows 128..159
  }
  for (int idx = tid; idx < 64*64; idx += 256) {
    const int n = idx & 63, k = idx >> 6;
    w2T[n*64 + k] = (bf16_t)w2[k*64 + n];
    w3T[n*64 + k] = (bf16_t)w3[k*64 + n];
  }
  for (int idx = tid; idx < NN*3; idx += 256) posS[idx] = pos[b*NN*3 + idx];
  maskS[tid] = mask[((size_t)(b*NN + i))*NN + tid];
  if (tid < 64) {
    sS[tid]  = s_in[(b*NN + i)*64 + tid];
    b1S[tid] = b1[tid]; b2S[tid] = b2[tid]; b3S[tid] = b3[tid];
    gS[tid]  = g[tid];  beS[tid] = be[tid];
  }
  __syncthreads();

  const int wave = tid >> 5;
  const int lane = tid & 31;
  const int lm   = lane & 15;   // N index (B/C/D) and M index (A) within tile
  const int hlf  = lane >> 4;   // lane half selects K-offset (A/B) / M-offset (C/D)
  const float pix = posS[i*3+0], piy = posS[i*3+1], piz = posS[i*3+2];
  bf16_t* ht = &htile[wave][0];

  for (int jt = wave; jt < 16; jt += 8) {
    const int j0 = jt * 16;

    // --- build 16x32 RBF A-fragment in registers ---
    const int ja = j0 + lm;
    const float mk = maskS[ja];
    const float dx = posS[ja*3+0]-pix, dy = posS[ja*3+1]-piy, dz = posS[ja*3+2]-piz;
    const float dist = sqrtf(dx*dx + dy*dy + dz*dz + 1e-10f) * mk;
    v16bf arf;
    #pragma unroll
    for (int e = 0; e < 16; ++e) {
      const int k = (hlf << 4) + e;  // A-layout K index
      const float tt = (dist - (float)k * (20.0f/31.0f)) * 1.6f; // sigma = 0.625
      arf[e] = (bf16_t)(__expf(-tt*tt) * mk);
    }

    // --- h1 = relu(rbf@Wr + s_i + t_j + b1) -> bf16 tile in LDS ---
    #pragma unroll
    for (int nt = 0; nt < 4; ++nt) {
      const int n = nt*16 + lm;
      const bf16_t* wb = &wrT[n*32 + hlf*16];
      const v16bf bfr = lds_frag(wb, wb + 8);
      v8f acc = {0.f,0.f,0.f,0.f,0.f,0.f,0.f,0.f};
      acc = __builtin_amdgcn_wmma_f32_16x16x32_bf16(false, arf, false, bfr,
                                                    (short)0, acc, false, false);
      const float sb = sS[n] + b1S[n];
      #pragma unroll
      for (int r = 0; r < 8; ++r) {
        const int mm = r + hlf*8;
        const float v = acc[r] + sb + t_in[((size_t)(b*NN + j0 + mm))*64 + n];
        ht[mm*64 + n] = (bf16_t)fmaxf(v, 0.f);
      }
    }

    // --- h2 = relu(h1@W2 + b2) -> bf16 tile (A loaded to regs before overwrite) ---
    {
      const bf16_t* ab = ht + lm*64 + hlf*8;
      const v16bf a0 = lds_frag(ab,      ab + 16);   // K = 0..31
      const v16bf a1 = lds_frag(ab + 32, ab + 48);   // K = 32..63
      #pragma unroll
      for (int nt = 0; nt < 4; ++nt) {
        const int n = nt*16 + lm;
        const bf16_t* wb = &w2T[n*64 + hlf*8];
        const v16bf bk0 = lds_frag(wb,      wb + 16);
        const v16bf bk1 = lds_frag(wb + 32, wb + 48);
        v8f acc = {0.f,0.f,0.f,0.f,0.f,0.f,0.f,0.f};
        acc = __builtin_amdgcn_wmma_f32_16x16x32_bf16(false, a0, false, bk0,
                                                      (short)0, acc, false, false);
        acc = __builtin_amdgcn_wmma_f32_16x16x32_bf16(false, a1, false, bk1,
                                                      (short)0, acc, false, false);
        const float bb = b2S[n];
        #pragma unroll
        for (int r = 0; r < 8; ++r) {
          const int mm = r + hlf*8;
          ht[mm*64 + n] = (bf16_t)fmaxf(acc[r] + bb, 0.f);
        }
      }
    }

    // --- h3 = h2@W3 + b3 (kept fp32 in registers) ---
    float vals[4][8];
    {
      const bf16_t* ab = ht + lm*64 + hlf*8;
      const v16bf a0 = lds_frag(ab,      ab + 16);
      const v16bf a1 = lds_frag(ab + 32, ab + 48);
      #pragma unroll
      for (int nt = 0; nt < 4; ++nt) {
        const int n = nt*16 + lm;
        const bf16_t* wb = &w3T[n*64 + hlf*8];
        const v16bf bk0 = lds_frag(wb,      wb + 16);
        const v16bf bk1 = lds_frag(wb + 32, wb + 48);
        v8f acc = {0.f,0.f,0.f,0.f,0.f,0.f,0.f,0.f};
        acc = __builtin_amdgcn_wmma_f32_16x16x32_bf16(false, a0, false, bk0,
                                                      (short)0, acc, false, false);
        acc = __builtin_amdgcn_wmma_f32_16x16x32_bf16(false, a1, false, bk1,
                                                      (short)0, acc, false, false);
        const float bb = b3S[n];
        #pragma unroll
        for (int r = 0; r < 8; ++r) vals[nt][r] = acc[r] + bb;
      }
    }

    // --- LayerNorm over E=64 per row via shfl_xor within 16-lane half-groups ---
    const size_t rowbase = ((size_t)(b*NN + i))*NN;
    #pragma unroll
    for (int r = 0; r < 8; ++r) {
      float sum = vals[0][r] + vals[1][r] + vals[2][r] + vals[3][r];
      #pragma unroll
      for (int o = 1; o < 16; o <<= 1) sum += __shfl_xor(sum, o, 32);
      const float mean = sum * (1.0f/64.0f);
      float q = 0.f;
      #pragma unroll
      for (int nt = 0; nt < 4; ++nt) { const float dv = vals[nt][r]-mean; q += dv*dv; }
      #pragma unroll
      for (int o = 1; o < 16; o <<= 1) q += __shfl_xor(q, o, 32);
      const float inv = rsqrtf(q*(1.0f/64.0f) + 1e-5f);
      const int mm = r + hlf*8;
      const size_t off = (rowbase + (size_t)(j0 + mm))*64;
      #pragma unroll
      for (int nt = 0; nt < 4; ++nt) {
        const int n = nt*16 + lm;
        edge_out[off + n] = (vals[nt][r]-mean)*inv*gS[n] + beS[n];
      }
    }
  }
}

// ---------------------------------------------------------------------------
// Kernel 3: agg[b,j,e] = (sum_i edge[b,i,j,e]) / (sum_i mask[b,i,j]+1e-10) * gate
// ---------------------------------------------------------------------------
__global__ __launch_bounds__(64) void k_agg(
    const float* __restrict__ edge, const float* __restrict__ mask,
    const int* __restrict__ atom, float* __restrict__ agg)
{
  const int bj = blockIdx.x;
  const int b = bj >> 8, j = bj & 255;
  const int e = threadIdx.x;
  float sum = 0.f;
  for (int i = 0; i < NN; ++i)
    sum += edge[(((size_t)(b*NN + i))*NN + j)*64 + e];
  float den = 1e-10f;
  for (int i = 0; i < NN; ++i)
    den += mask[((size_t)(b*NN + i))*NN + j];
  const float gate = (atom[b*NN + j] > 0) ? 1.f : 0.f;  // min(atom,1)
  agg[(b*NN + j)*64 + e] = sum / den * gate;
}

// ---------------------------------------------------------------------------
// Kernel 4: node MLP + residual: out = node + LN(relu(relu([node,agg]@W1)@W2)@W3)
// ---------------------------------------------------------------------------
__global__ __launch_bounds__(64) void k_nodemlp(
    const float* __restrict__ node, const float* __restrict__ agg,
    const float* __restrict__ w1, const float* __restrict__ b1,
    const float* __restrict__ w2, const float* __restrict__ b2,
    const float* __restrict__ w3, const float* __restrict__ b3,
    const float* __restrict__ g, const float* __restrict__ be,
    float* __restrict__ out_node)
{
  __shared__ float xb[128], hb[64], rb[64];
  const int row = blockIdx.x;
  const int d = threadIdx.x;
  const float n0 = node[row*64 + d];
  xb[d] = n0; xb[64 + d] = agg[row*64 + d];
  __syncthreads();
  float a1 = b1[d];
  for (int k = 0; k < 128; ++k) a1 += xb[k]*w1[k*64 + d];
  a1 = fmaxf(a1, 0.f);
  hb[d] = a1; __syncthreads();
  float a2 = b2[d];
  for (int k = 0; k < 64; ++k) a2 += hb[k]*w2[k*64 + d];
  a2 = fmaxf(a2, 0.f);
  __syncthreads(); hb[d] = a2; __syncthreads();
  float a3 = b3[d];
  for (int k = 0; k < 64; ++k) a3 += hb[k]*w3[k*64 + d];
  rb[d] = a3; __syncthreads();
  for (int o = 32; o > 0; o >>= 1) { if (d < o) rb[d] += rb[d+o]; __syncthreads(); }
  const float mean = rb[0] * (1.0f/64.0f);
  __syncthreads();
  const float dv = a3 - mean;
  rb[d] = dv*dv; __syncthreads();
  for (int o = 32; o > 0; o >>= 1) { if (d < o) rb[d] += rb[d+o]; __syncthreads(); }
  const float inv = rsqrtf(rb[0]*(1.0f/64.0f) + 1e-5f);
  out_node[row*64 + d] = n0 + dv*inv*g[d] + be[d];
}

// ---------------------------------------------------------------------------
extern "C" void kernel_launch(void* const* d_in, const int* in_sizes, int n_in,
                              void* d_out, int out_size, void* d_ws, size_t ws_size,
                              hipStream_t stream) {
  const int*   atom    = (const int*)  d_in[0];
  const float* pos     = (const float*)d_in[1];
  const float* mask    = (const float*)d_in[2];
  const float* emb     = (const float*)d_in[3];
  const float* ne_w    = (const float*)d_in[4];
  const float* ne_b    = (const float*)d_in[5];
  const float* ne_g    = (const float*)d_in[6];
  const float* ne_beta = (const float*)d_in[7];
  const float* ee_w1   = (const float*)d_in[8];
  const float* ee_b1   = (const float*)d_in[9];
  const float* ee_w2   = (const float*)d_in[10];
  const float* ee_b2   = (const float*)d_in[11];
  const float* ee_w3   = (const float*)d_in[12];
  const float* ee_b3   = (const float*)d_in[13];
  const float* ee_g    = (const float*)d_in[14];
  const float* ee_beta = (const float*)d_in[15];
  const float* na_w1   = (const float*)d_in[16];
  const float* na_b1   = (const float*)d_in[17];
  const float* na_w2   = (const float*)d_in[18];
  const float* na_b2   = (const float*)d_in[19];
  const float* na_w3   = (const float*)d_in[20];
  const float* na_b3   = (const float*)d_in[21];
  const float* na_g    = (const float*)d_in[22];
  const float* na_beta = (const float*)d_in[23];

  const int rows = NB*NN;                 // 2048
  float* node = (float*)d_ws;             // rows*64
  float* s    = node + rows*64;
  float* t    = s    + rows*64;
  float* agg  = t    + rows*64;

  float* out_node = (float*)d_out;                 // (B,N,D)
  float* out_edge = out_node + rows*64;            // (B,N,N,E)

  k_node   <<<rows, 64,  0, stream>>>(atom, emb, ne_w, ne_b, ne_g, ne_beta,
                                      ee_w1, node, s, t);
  k_edge   <<<rows, 256, 0, stream>>>(pos, mask, ee_w1, ee_b1, ee_w2, ee_b2,
                                      ee_w3, ee_b3, ee_g, ee_beta, s, t, out_edge);
  k_agg    <<<rows, 64,  0, stream>>>(out_edge, mask, atom, agg);
  k_nodemlp<<<rows, 64,  0, stream>>>(node, agg, na_w1, na_b1, na_w2, na_b2,
                                      na_w3, na_b3, na_g, na_beta, out_node);
}
